// multi_head_attention_layer_84146999263581
// MI455X (gfx1250) — matrix operand order repaired and checked by reading, not compiled, not run on hardware
//
#include <hip/hip_runtime.h>
#include <hip/hip_bf16.h>

// MHA for MI455X (gfx1250): bf16 WMMA everywhere, f32 accumulate.
// Pipeline: fp32->bf16 convert -> 3 projection GEMMs (WMMA, async-to-LDS staging) ->
//           flash attention (WMMA QK^T + streaming softmax + WMMA PV) ->
//           output GEMM fused with final mask (WMMA), fp32 out.

typedef __attribute__((ext_vector_type(16))) __bf16 v16bf;
typedef __attribute__((ext_vector_type(8)))  float  v8f;

__device__ __forceinline__ v16bf mk_frag(const __bf16* p0, const __bf16* p1) {
    union { uint4 q[2]; v16bf v; } u;
    u.q[0] = *(const uint4*)p0;
    u.q[1] = *(const uint4*)p1;
    return u.v;
}

// CDNA5 async DMA: global -> LDS, 16B per lane, tracked by ASYNCcnt.
// lds_off: low 32 bits of the flat shared-aperture address == LDS byte address.
__device__ __forceinline__ void async_copy_b128(const void* lds_dst, const void* gsrc) {
    unsigned lds_off = (unsigned)(unsigned long long)lds_dst;
    asm volatile("global_load_async_to_lds_b128 %0, %1, off"
                 :: "v"(lds_off), "v"(gsrc) : "memory");
}
__device__ __forceinline__ void wait_async0() {
    asm volatile("s_wait_asynccnt 0" ::: "memory");
}

__global__ __launch_bounds__(256) void cvt_f32_bf16(const float* __restrict__ in,
                                                    __bf16* __restrict__ out, int n) {
    int i = blockIdx.x * blockDim.x + threadIdx.x;
    if (i < n) out[i] = (__bf16)in[i];
}

// C = X(M x K) @ W(N x K)^T (+bias). mode 0: write bf16 to per-head (n,s,d) layout.
// mode 1: write fp32 to (b,s,1024) with mask(s<seq_len[b]) applied.
__global__ __launch_bounds__(256) void gemm_bf16_wmma(
    const __bf16* __restrict__ X, const __bf16* __restrict__ W,
    const float* __restrict__ bias, void* __restrict__ out,
    const long long* __restrict__ seq_len, int Kdim, int mode)
{
    __shared__ __align__(16) __bf16 ldsA[128 * 32];
    __shared__ __align__(16) __bf16 ldsB[128 * 32];

    const int tid  = threadIdx.x;
    const int wid  = tid >> 5;
    const int lane = tid & 31;
    const int hf   = lane >> 4;    // half of wave (C-layout row offset 8*hf)
    const int lm   = lane & 15;    // column within tile
    const int m0   = blockIdx.x * 128;
    const int n0   = blockIdx.y * 128;
    const int wm   = (wid & 3) * 32;   // wave M offset within block
    const int wn   = (wid >> 2) * 64;  // wave N offset within block

    v8f acc[2][4];
#pragma unroll
    for (int tm = 0; tm < 2; ++tm)
#pragma unroll
        for (int tn = 0; tn < 4; ++tn)
#pragma unroll
            for (int i = 0; i < 8; ++i) acc[tm][tn][i] = 0.0f;

    const int ksteps = Kdim >> 5;
    for (int ks = 0; ks < ksteps; ++ks) {
        const int k0 = ks << 5;
        __syncthreads();   // previous iteration's fragment reads are done
        // Stage A/B tiles (128x32 bf16 each) via async DMA: no VGPR round-trip,
        // no per-load s_wait_loadcnt chain.
#pragma unroll
        for (int i = 0; i < 2; ++i) {
            int idx = tid * 16 + i * 8;          // 8 bf16 = 16B per copy
            int r = idx >> 5, c = idx & 31;
            async_copy_b128(&ldsA[idx], &X[(size_t)(m0 + r) * Kdim + k0 + c]);
            async_copy_b128(&ldsB[idx], &W[(size_t)(n0 + r) * Kdim + k0 + c]);
        }
        if (ks + 1 < ksteps) { // warm L2 for next K slab (global_prefetch_b8)
            __builtin_prefetch(&X[(size_t)(m0 + (tid >> 1)) * Kdim + k0 + 32], 0, 1);
            __builtin_prefetch(&W[(size_t)(n0 + (tid >> 1)) * Kdim + k0 + 32], 0, 1);
        }
        wait_async0();     // all of this thread's DMAs have landed in LDS
        __syncthreads();   // ...and everyone else's too

        v16bf af[2], bfr[4];
#pragma unroll
        for (int tm = 0; tm < 2; ++tm) {
            int r = wm + tm * 16 + lm;            // A: row per lane, K split by half
            af[tm] = mk_frag(&ldsA[r * 32 + hf * 8], &ldsA[r * 32 + 16 + hf * 8]);
        }
#pragma unroll
        for (int tn = 0; tn < 4; ++tn) {
            int r = wn + tn * 16 + lm;            // B: column per lane = weight row
            bfr[tn] = mk_frag(&ldsB[r * 32 + hf * 8], &ldsB[r * 32 + 16 + hf * 8]);
        }
#pragma unroll
        for (int tm = 0; tm < 2; ++tm)
#pragma unroll
            for (int tn = 0; tn < 4; ++tn)
                acc[tm][tn] = __builtin_amdgcn_wmma_f32_16x16x32_bf16(
                    false, af[tm], false, bfr[tn], (short)0, acc[tm][tn], false, false);
    }

#pragma unroll
    for (int tm = 0; tm < 2; ++tm)
#pragma unroll
        for (int tn = 0; tn < 4; ++tn)
#pragma unroll
            for (int i = 0; i < 8; ++i) {
                int gm = m0 + wm + tm * 16 + i + hf * 8;
                int gn = n0 + wn + tn * 16 + lm;
                float v = acc[tm][tn][i];
                if (bias) v += bias[gn];
                int bb = gm >> 11, ss = gm & 2047;
                if (mode == 0) {
                    int h = gn >> 6, d = gn & 63;
                    ((__bf16*)out)[(((size_t)(bb * 16 + h)) * 2048 + ss) * 64 + d] = (__bf16)v;
                } else {
                    float mval = (ss < (int)seq_len[bb]) ? 1.0f : 0.0f;
                    ((float*)out)[(size_t)gm * 1024 + gn] = v * mval;
                }
            }
}

// Flash attention per (128-query-row block, head-batch n). Zero-fill masking
// (reference uses masked_fill(0), NOT -inf) with streaming softmax.
__global__ __launch_bounds__(256) void attn_wmma(
    const __bf16* __restrict__ Q, const __bf16* __restrict__ Km,
    const __bf16* __restrict__ V, __bf16* __restrict__ Cb,
    const long long* __restrict__ seq_len)
{
    __shared__ __align__(16) __bf16 ldsVt[64 * 32];     // V chunk transposed (d, kpos)
    __shared__ __align__(16) __bf16 ldsP[8 * 16 * 32];  // per-wave P tiles

    const int tid  = threadIdx.x;
    const int wid  = tid >> 5;
    const int lane = tid & 31;
    const int hf   = lane >> 4;
    const int lm   = lane & 15;
    const int n    = blockIdx.y;
    const int b    = n >> 4;
    const int h    = n & 15;
    const int L    = (int)seq_len[b];
    const int qbase = blockIdx.x * 128 + wid * 16;
    const int ha   = hf * 8;

    // Q A-fragments: row = qbase+lm, contract d=64 over two K-steps of 32
    const __bf16* qrow = Q + ((size_t)n * 2048 + (qbase + lm)) * 64;
    v16bf aq0 = mk_frag(qrow + ha,      qrow + 16 + ha);
    v16bf aq1 = mk_frag(qrow + 32 + ha, qrow + 48 + ha);

    v8f ctx[4];
    float rowm[8], rowl[8];
#pragma unroll
    for (int tn = 0; tn < 4; ++tn)
#pragma unroll
        for (int i = 0; i < 8; ++i) ctx[tn][i] = 0.0f;
#pragma unroll
    for (int i = 0; i < 8; ++i) { rowm[i] = -3.0e38f; rowl[i] = 0.0f; }

    __bf16* myP = &ldsP[wid * 512];
    const float scl = 0.03125f; // 1/sqrt(1024)

    for (int kc = 0; kc < 64; ++kc) {
        const int kpos0 = kc * 32;

        // stage V chunk (32 x 64) transposed into LDS (64 x 32)
        __syncthreads();
        {
            int idx = tid * 8;
            int kr = idx >> 6;     // key row in chunk
            int dc = idx & 63;     // d column base
            uint4 raw = *(const uint4*)&V[((size_t)n * 2048 + kpos0 + kr) * 64 + dc];
            const __bf16* e = (const __bf16*)&raw;
#pragma unroll
            for (int j2 = 0; j2 < 8; ++j2) ldsVt[(dc + j2) * 32 + kr] = e[j2];
        }
        __syncthreads();

        // scores: 16 rows x 32 key columns (two 16-col WMMA tiles, K=64 in 2 steps)
        v8f sc[2];
#pragma unroll
        for (int j = 0; j < 2; ++j) {
#pragma unroll
            for (int i = 0; i < 8; ++i) sc[j][i] = 0.0f;
            const __bf16* krow = Km + ((size_t)n * 2048 + (kpos0 + j * 16 + lm)) * 64;
            v16bf bk0 = mk_frag(krow + ha,      krow + 16 + ha);
            v16bf bk1 = mk_frag(krow + 32 + ha, krow + 48 + ha);
            sc[j] = __builtin_amdgcn_wmma_f32_16x16x32_bf16(false, aq0, false, bk0,
                                                            (short)0, sc[j], false, false);
            sc[j] = __builtin_amdgcn_wmma_f32_16x16x32_bf16(false, aq1, false, bk1,
                                                            (short)0, sc[j], false, false);
        }

        // scale + zero-fill masking (column index is lane-uniform per tile)
        bool keep0 = (kpos0 + lm) < L;
        bool keep1 = (kpos0 + 16 + lm) < L;
#pragma unroll
        for (int i = 0; i < 8; ++i) {
            sc[0][i] = keep0 ? sc[0][i] * scl : 0.0f;
            sc[1][i] = keep1 ? sc[1][i] * scl : 0.0f;
        }

        // streaming softmax per row (reduce across the 16-lane half)
#pragma unroll
        for (int i = 0; i < 8; ++i) {
            float loc = fmaxf(sc[0][i], sc[1][i]);
#pragma unroll
            for (int off = 1; off < 16; off <<= 1)
                loc = fmaxf(loc, __shfl_xor(loc, off, 32));
            float mnew = fmaxf(rowm[i], loc);
            float p0 = __expf(sc[0][i] - mnew);
            float p1 = __expf(sc[1][i] - mnew);
            float ps = p0 + p1;
#pragma unroll
            for (int off = 1; off < 16; off <<= 1)
                ps += __shfl_xor(ps, off, 32);
            float corr = __expf(rowm[i] - mnew);
            rowl[i] = rowl[i] * corr + ps;
            rowm[i] = mnew;
#pragma unroll
            for (int tn = 0; tn < 4; ++tn) ctx[tn][i] *= corr;
            int prow = i + hf * 8;                   // C-layout -> LDS P tile
            myP[prow * 32 + lm]      = (__bf16)p0;
            myP[prow * 32 + 16 + lm] = (__bf16)p1;
        }
        // cross-lane LDS dependency within the wave: wait for the P stores
        asm volatile("s_wait_dscnt 0" ::: "memory");

        // PV: P (16x32) @ V (32x64) accumulated into ctx
        v16bf ap = mk_frag(&myP[lm * 32 + ha], &myP[lm * 32 + 16 + ha]);
#pragma unroll
        for (int tn = 0; tn < 4; ++tn) {
            const __bf16* vcol = &ldsVt[(tn * 16 + lm) * 32 + ha];
            v16bf bv = mk_frag(vcol, vcol + 16);
            ctx[tn] = __builtin_amdgcn_wmma_f32_16x16x32_bf16(false, ap, false, bv,
                                                              (short)0, ctx[tn], false, false);
        }
    }

    // normalize and write context in (b, s, h*64+d) layout for the final GEMM
#pragma unroll
    for (int tn = 0; tn < 4; ++tn)
#pragma unroll
        for (int i = 0; i < 8; ++i) {
            int s  = qbase + i + hf * 8;
            int dm = h * 64 + tn * 16 + lm;
            Cb[((size_t)b * 2048 + s) * 1024 + dm] = (__bf16)(ctx[tn][i] / rowl[i]);
        }
}

extern "C" void kernel_launch(void* const* d_in, const int* in_sizes, int n_in,
                              void* d_out, int out_size, void* d_ws, size_t ws_size,
                              hipStream_t stream)
{
    const float*     query   = (const float*)d_in[0];
    const long long* seq_len = (const long long*)d_in[1];  // int64 in reference
    const float*     wq      = (const float*)d_in[2];
    const float*     bq      = (const float*)d_in[3];
    const float*     wk      = (const float*)d_in[4];
    const float*     bk      = (const float*)d_in[5];
    const float*     wv      = (const float*)d_in[6];
    const float*     bv      = (const float*)d_in[7];
    const float*     wg      = (const float*)d_in[8];

    char* ws = (char*)d_ws;
    const size_t MB = 1024 * 1024;
    __bf16* Xb  = (__bf16*)(ws + 0);        // bf16 query (also reused as context Cb)
    __bf16* Qb  = (__bf16*)(ws + 16 * MB);  // (n=b*16+h, s, 64)
    __bf16* Kb  = (__bf16*)(ws + 32 * MB);
    __bf16* Vb  = (__bf16*)(ws + 48 * MB);
    __bf16* Wqb = (__bf16*)(ws + 64 * MB);  // 2 MB each
    __bf16* Wkb = (__bf16*)(ws + 66 * MB);
    __bf16* Wvb = (__bf16*)(ws + 68 * MB);
    __bf16* Wgb = (__bf16*)(ws + 70 * MB);
    __bf16* Cb  = Xb;                       // X dead after projections

    const int NQ = 4 * 2048 * 1024;
    const int NW = 1024 * 1024;
    cvt_f32_bf16<<<(NQ + 255) / 256, 256, 0, stream>>>(query, Xb, NQ);
    cvt_f32_bf16<<<(NW + 255) / 256, 256, 0, stream>>>(wq, Wqb, NW);
    cvt_f32_bf16<<<(NW + 255) / 256, 256, 0, stream>>>(wk, Wkb, NW);
    cvt_f32_bf16<<<(NW + 255) / 256, 256, 0, stream>>>(wv, Wvb, NW);
    cvt_f32_bf16<<<(NW + 255) / 256, 256, 0, stream>>>(wg, Wgb, NW);

    dim3 ggrid(64, 8); // 8192/128 x 1024/128
    gemm_bf16_wmma<<<ggrid, 256, 0, stream>>>(Xb, Wqb, bq, (void*)Qb, nullptr, 1024, 0);
    gemm_bf16_wmma<<<ggrid, 256, 0, stream>>>(Xb, Wkb, bk, (void*)Kb, nullptr, 1024, 0);
    gemm_bf16_wmma<<<ggrid, 256, 0, stream>>>(Xb, Wvb, bv, (void*)Vb, nullptr, 1024, 0);

    attn_wmma<<<dim3(16, 64), 256, 0, stream>>>(Qb, Kb, Vb, Cb, seq_len);

    gemm_bf16_wmma<<<ggrid, 256, 0, stream>>>(Cb, Wgb, nullptr, d_out, seq_len, 1024, 1);
}
